// LocalBlock_81758997447240
// MI455X (gfx1250) — matrix operand order repaired and checked by reading, not compiled, not run on hardware
//
#include <hip/hip_runtime.h>
#include <hip/hip_bf16.h>
#include <math.h>

// ---------------- problem constants ----------------
#define BB   2
#define NN   8192
#define IC   64
#define OC   128
#define KNN  16
#define MM   (BB * NN)          // 16384 rows
#define EPSF 1e-5f
#define PTILE 2048              // knn pts LDS tile (points)

typedef __attribute__((ext_vector_type(16))) _Float16 v16h;
typedef __attribute__((ext_vector_type(8)))  float    v8f;
typedef __attribute__((ext_vector_type(4)))  int      v4i;

__device__ __forceinline__ float gelu_exact(float v) {
    return 0.5f * v * (1.0f + erff(v * 0.70710678118654752f));
}

// A-matrix 16x32 f16 fragment (ISA 7.12.2): lane r=lane&15 holds row r.
// h=lane>>4: VGPR0-3 hold K = 0..7 (+8h), VGPR4-7 hold K = 16..23 (+8h).
__device__ __forceinline__ v16h load_a_frag(const _Float16* __restrict__ base,
                                            int lda, int kblock, int lane) {
    const int r = lane & 15, h = lane >> 4;
    const _Float16* row = base + r * lda + kblock;
    v16h a;
#pragma unroll
    for (int i = 0; i < 8; ++i) {
        const int k0 = (i < 4 ? 2 * i : 16 + 2 * (i - 4)) + 8 * h;
        a[2 * i]     = row[k0];
        a[2 * i + 1] = row[k0 + 1];
    }
    return a;
}

// Pre-packed B fragment: one contiguous 32B vector per (n-tile, k-tile, lane).
__device__ __forceinline__ v16h load_b_packed(const _Float16* __restrict__ packed,
                                              int nkt, int kt, int ntile, int lane) {
    return ((const v16h*)packed)[(size_t)(ntile * nkt + kt) * 32 + lane];
}

// Pack weight W (K x 128, row-major unless transpose) into B-fragment order.
__device__ __forceinline__ void pack_b(int i, int K, const float* __restrict__ W,
                                       bool transpose, _Float16* __restrict__ out) {
    const int e    = i & 15;
    const int l    = (i >> 4) & 31;
    const int rest = i >> 9;
    const int nkt  = K >> 5;
    const int kt   = rest % nkt;
    const int nt   = rest / nkt;
    const int k    = kt * 32 + (e & 7) + 8 * (l >> 4) + 16 * (e >> 3);
    const int col  = nt * 16 + (l & 15);
    const float v  = transpose ? W[col * K + k] : W[k * OC + col];
    out[i] = (_Float16)v;
}

// ---------------- kernel 0: pack weights to f16 fragments, zero BN acc ----
__global__ __launch_bounds__(256) void convert_zero_kernel(
    const float* __restrict__ feats, const float* __restrict__ fc1_w,
    const float* __restrict__ fc2_w, const float* __restrict__ pw_w,
    const float* __restrict__ proj_w,
    _Float16* __restrict__ feats16, _Float16* __restrict__ w1p,
    _Float16* __restrict__ w2p, _Float16* __restrict__ wpwp,
    _Float16* __restrict__ wprojp, float* __restrict__ gacc /* 256 floats */) {
    const int i = blockIdx.x * 256 + threadIdx.x;
    if (i < MM * IC)  feats16[i] = (_Float16)feats[i];
    if (i < IC * OC)  pack_b(i, IC, fc1_w,  false, w1p);
    if (i < OC * OC)  pack_b(i, OC, fc2_w,  false, w2p);
    if (i < OC * OC)  pack_b(i, OC, pw_w,   true,  wpwp);  // y = x @ pw_w.T
    if (i < IC * OC)  pack_b(i, IC, proj_w, false, wprojp);
    if (i < 256)      gacc[i] = 0.0f;
}

// ---------------- kernel 1: KNN + neighbor feature mean ----------------
// One wave per query. pts staged through LDS tiles (async-to-LDS if available),
// per-lane streaming top-16, LDS merge, fused 16-neighbor feature mean.
__global__ __launch_bounds__(256) void knn_gather_kernel(
    const float* __restrict__ pts, const float* __restrict__ feats,
    _Float16* __restrict__ x16) {
    __shared__ float4 spts4[PTILE * 3 / 4];           // 24 KB point tile
    __shared__ float  sd[8][512];
    __shared__ int    si[8][512];
    __shared__ int    sel[8][KNN];
    float* spts = (float*)spts4;

    const int tid = threadIdx.x, lane = tid & 31, w = tid >> 5;
    const int qid = blockIdx.x * 8 + w;
    const int b = qid >> 13, n = qid & (NN - 1);
    const float* __restrict__ pb = pts + (size_t)b * NN * 3;
    __builtin_prefetch(pb, 0, 0);

    const float qx = pb[n * 3 + 0], qy = pb[n * 3 + 1], qz = pb[n * 3 + 2];

    float bd[KNN]; int bi[KNN];
#pragma unroll
    for (int i = 0; i < KNN; ++i) { bd[i] = 3.4e38f; bi[i] = 0; }
    float worst = 3.4e38f;

    for (int t = 0; t < NN / PTILE; ++t) {
        const float* gsrc = pb + (size_t)t * (PTILE * 3);
#if __has_builtin(__builtin_amdgcn_global_load_async_to_lds_b128)
        {
            // builtin expects (int4 addrspace(1)*, int4 addrspace(3)*, imm offset, imm cpol)
            __attribute__((address_space(1))) v4i* gp =
                (__attribute__((address_space(1))) v4i*)(float*)gsrc;
            __attribute__((address_space(3))) v4i* lp =
                (__attribute__((address_space(3))) v4i*)spts;
#pragma unroll
            for (int q = 0; q < (PTILE * 3 * 4) / (256 * 16); ++q) {   // 6 x b128
                const int idx = q * 256 + tid;
                __builtin_amdgcn_global_load_async_to_lds_b128(gp + idx, lp + idx, 0, 0);
            }
#if __has_builtin(__builtin_amdgcn_s_wait_asynccnt)
            __builtin_amdgcn_s_wait_asynccnt(0);
#else
            asm volatile("s_wait_asynccnt 0x0" ::: "memory");
#endif
        }
#else
#pragma unroll
        for (int q = 0; q < (PTILE * 3) / (256 * 4); ++q) {
            const int off = q * 256 + tid;
            spts4[off] = ((const float4*)gsrc)[off];
        }
#endif
        __syncthreads();

        for (int j = lane; j < PTILE; j += 32) {
            const float dx = spts[j * 3 + 0] - qx;
            const float dy = spts[j * 3 + 1] - qy;
            const float dz = spts[j * 3 + 2] - qz;
            const float d2 = dx * dx + dy * dy + dz * dz;
            if (d2 < worst) {
                const int jj = t * PTILE + j;
                int am = 0; float mv = bd[0];
#pragma unroll
                for (int i = 1; i < KNN; ++i) if (bd[i] > mv) { mv = bd[i]; am = i; }
#pragma unroll
                for (int i = 0; i < KNN; ++i) if (i == am) { bd[i] = d2; bi[i] = jj; }
                float nw = bd[0];
#pragma unroll
                for (int i = 1; i < KNN; ++i) nw = fmaxf(nw, bd[i]);
                worst = nw;
            }
        }
        __syncthreads();   // tile consumed before overwrite
    }

#pragma unroll
    for (int i = 0; i < KNN; ++i) {
        sd[w][lane * KNN + i] = bd[i];
        si[w][lane * KNN + i] = bi[i];
    }
    __syncthreads();

    // 16 rounds of cooperative global-min over the 512 wave candidates
    for (int round = 0; round < KNN; ++round) {
        float m = 3.4e38f; int mpos = 0;
#pragma unroll
        for (int i = 0; i < 16; ++i) {
            const int p = i * 32 + lane;
            const float v = sd[w][p];
            if (v < m) { m = v; mpos = p; }
        }
#pragma unroll
        for (int off = 16; off > 0; off >>= 1) {
            const float om = __shfl_xor(m, off, 32);
            const int   op = __shfl_xor(mpos, off, 32);
            if (om < m || (om == m && op < mpos)) { m = om; mpos = op; }
        }
        if (lane == 0) { sel[w][round] = si[w][mpos]; sd[w][mpos] = 3.4e38f; }
        __syncthreads();
    }

    // gather-mean of the 16 selected neighbors: lane owns channels lane, lane+32
    const float* __restrict__ fb = feats + (size_t)b * NN * IC;
    float a0 = 0.0f, a1 = 0.0f;
#pragma unroll
    for (int t = 0; t < KNN; ++t) {
        const int id = sel[w][t];
        a0 += fb[(size_t)id * IC + lane];
        a1 += fb[(size_t)id * IC + lane + 32];
    }
    _Float16* xr = x16 + (size_t)qid * IC;
    xr[lane]      = (_Float16)(a0 * (1.0f / 16.0f));
    xr[lane + 32] = (_Float16)(a1 * (1.0f / 16.0f));
}

// ---------------- kernel 2: fused MLP over 16-row stripes ----------------
__global__ __launch_bounds__(256) void mlp_kernel(
    const _Float16* __restrict__ x16,  // (M,64)
    const _Float16* __restrict__ w1p,  // packed fragments
    const float* __restrict__ fc1_b,
    const float* __restrict__ ln1_g, const float* __restrict__ ln1_b,
    const _Float16* __restrict__ w2p,
    const float* __restrict__ fc2_b,
    const float* __restrict__ ln2_g, const float* __restrict__ ln2_b,
    const float* __restrict__ dw_w, const float* __restrict__ dw_b,
    const _Float16* __restrict__ wpwp,
    const float* __restrict__ pw_b,
    float* __restrict__ y_out,         // (M,128)
    float* __restrict__ gsum, float* __restrict__ gsumsq) {
    __shared__ float    t32[16 * OC];
    __shared__ _Float16 t16[16 * OC];
    __shared__ float    psum[256], psumsq[256];
    __shared__ float    rmean[16], rrstd[16];

    const int tid = threadIdx.x, lane = tid & 31, w = tid >> 5;
    const int r = lane & 15, h = lane >> 4;
    const int n0 = w * 16;
    const int m0 = blockIdx.x * 16;
    const int row = tid >> 4, seg = tid & 15;   // LN work split

    // stage x tile (16x64 f16 = 2KB) into LDS with vector copies
    if (tid < 128)
        ((float4*)t16)[tid] = ((const float4*)(x16 + (size_t)m0 * IC))[tid];
    __syncthreads();

    // ---- FC1: (16x64) @ (64x128), 2 WMMA per wave ----
    v8f acc;
    {
        const float bias = fc1_b[n0 + r];
#pragma unroll
        for (int j = 0; j < 8; ++j) acc[j] = bias;
#pragma unroll
        for (int kt = 0; kt < IC / 32; ++kt) {
            const v16h a = load_a_frag(&t16[0], IC, kt * 32, lane);
            const v16h b = load_b_packed(w1p, IC / 32, kt, w, lane);
            acc = __builtin_amdgcn_wmma_f32_16x16x32_f16(false, a, false, b,
                                                         (short)0, acc, false, false);
        }
#pragma unroll
        for (int j = 0; j < 8; ++j)
            t32[(j + 8 * h) * OC + n0 + r] = gelu_exact(acc[j]);
    }
    __syncthreads();

    // ---- LN1 -> t16 ----
    {
        float s = 0.0f, s2 = 0.0f;
#pragma unroll
        for (int c = 0; c < 8; ++c) {
            const float v = t32[row * OC + seg * 8 + c];
            s += v; s2 += v * v;
        }
        psum[row * 16 + seg] = s; psumsq[row * 16 + seg] = s2;
        __syncthreads();
        if (tid < 16) {
            float S = 0.0f, S2 = 0.0f;
#pragma unroll
            for (int i = 0; i < 16; ++i) { S += psum[tid * 16 + i]; S2 += psumsq[tid * 16 + i]; }
            const float mu = S * (1.0f / OC);
            const float var = S2 * (1.0f / OC) - mu * mu;
            rmean[tid] = mu; rrstd[tid] = rsqrtf(var + EPSF);
        }
        __syncthreads();
        const float mu = rmean[row], rs = rrstd[row];
#pragma unroll
        for (int c = 0; c < 8; ++c) {
            const int col = seg * 8 + c;
            const float v = (t32[row * OC + col] - mu) * rs * ln1_g[col] + ln1_b[col];
            t16[row * OC + col] = (_Float16)v;
        }
    }
    __syncthreads();

    // ---- FC2: (16x128) @ (128x128), 4 WMMA per wave ----
    {
        const float bias = fc2_b[n0 + r];
#pragma unroll
        for (int j = 0; j < 8; ++j) acc[j] = bias;
#pragma unroll
        for (int kt = 0; kt < OC / 32; ++kt) {
            const v16h a = load_a_frag(&t16[0], OC, kt * 32, lane);
            const v16h b = load_b_packed(w2p, OC / 32, kt, w, lane);
            acc = __builtin_amdgcn_wmma_f32_16x16x32_f16(false, a, false, b,
                                                         (short)0, acc, false, false);
        }
    }
    __syncthreads();   // everyone done reading t16 before overwrite
#pragma unroll
    for (int j = 0; j < 8; ++j)
        t32[(j + 8 * h) * OC + n0 + r] = gelu_exact(acc[j]);
    __syncthreads();

    // ---- LN2 fused with 1x1 depthwise affine -> t16 ----
    {
        float s = 0.0f, s2 = 0.0f;
#pragma unroll
        for (int c = 0; c < 8; ++c) {
            const float v = t32[row * OC + seg * 8 + c];
            s += v; s2 += v * v;
        }
        psum[row * 16 + seg] = s; psumsq[row * 16 + seg] = s2;
        __syncthreads();
        if (tid < 16) {
            float S = 0.0f, S2 = 0.0f;
#pragma unroll
            for (int i = 0; i < 16; ++i) { S += psum[tid * 16 + i]; S2 += psumsq[tid * 16 + i]; }
            const float mu = S * (1.0f / OC);
            const float var = S2 * (1.0f / OC) - mu * mu;
            rmean[tid] = mu; rrstd[tid] = rsqrtf(var + EPSF);
        }
        __syncthreads();
        const float mu = rmean[row], rs = rrstd[row];
#pragma unroll
        for (int c = 0; c < 8; ++c) {
            const int col = seg * 8 + c;
            float v = (t32[row * OC + col] - mu) * rs * ln2_g[col] + ln2_b[col];
            v = v * dw_w[col] + dw_b[col];          // depthwise 1x1
            t16[row * OC + col] = (_Float16)v;
        }
    }
    __syncthreads();

    // ---- PW: (16x128) @ (128x128), 4 WMMA per wave ----
    {
        const float bias = pw_b[n0 + r];
#pragma unroll
        for (int j = 0; j < 8; ++j) acc[j] = bias;
#pragma unroll
        for (int kt = 0; kt < OC / 32; ++kt) {
            const v16h a = load_a_frag(&t16[0], OC, kt * 32, lane);
            const v16h b = load_b_packed(wpwp, OC / 32, kt, w, lane);
            acc = __builtin_amdgcn_wmma_f32_16x16x32_f16(false, a, false, b,
                                                         (short)0, acc, false, false);
        }
    }
    __syncthreads();   // t32 free to reuse
#pragma unroll
    for (int j = 0; j < 8; ++j) {
        const int rr = j + 8 * h, col = n0 + r;
        const float v = acc[j];
        y_out[(size_t)(m0 + rr) * OC + col] = v;
        t32[rr * OC + col] = v;
    }
    __syncthreads();

    // ---- BN partial sums: one atomicAdd per channel per block ----
    if (tid < OC) {
        float s = 0.0f, s2 = 0.0f;
#pragma unroll
        for (int rw = 0; rw < 16; ++rw) {
            const float v = t32[rw * OC + tid];
            s += v; s2 += v * v;
        }
        atomicAdd(&gsum[tid], s);
        atomicAdd(&gsumsq[tid], s2);
    }
}

// ---------------- kernel 3: BN finalize + GELU + residual proj ----------------
__global__ __launch_bounds__(256) void finalize_kernel(
    const _Float16* __restrict__ feats16,  // (M,64)
    const _Float16* __restrict__ wprojp,   // packed fragments
    const float* __restrict__ proj_b,
    const float* __restrict__ bn_g, const float* __restrict__ bn_b,
    const float* __restrict__ y,           // (M,128)
    const float* __restrict__ gsum, const float* __restrict__ gsumsq,
    float* __restrict__ out) {
    __shared__ _Float16 a16[16 * IC];

    const int tid = threadIdx.x, lane = tid & 31, w = tid >> 5;
    const int r = lane & 15, h = lane >> 4;
    const int n0 = w * 16;
    const int m0 = blockIdx.x * 16;

    if (tid < 128)
        ((float4*)a16)[tid] = ((const float4*)(feats16 + (size_t)m0 * IC))[tid];
    __syncthreads();

    v8f acc;
    const float bias = proj_b[n0 + r];
#pragma unroll
    for (int j = 0; j < 8; ++j) acc[j] = bias;
#pragma unroll
    for (int kt = 0; kt < IC / 32; ++kt) {
        const v16h a = load_a_frag(&a16[0], IC, kt * 32, lane);
        const v16h b = load_b_packed(wprojp, IC / 32, kt, w, lane);
        acc = __builtin_amdgcn_wmma_f32_16x16x32_f16(false, a, false, b,
                                                     (short)0, acc, false, false);
    }

    const int col = n0 + r;
    const float inv_m = 1.0f / (float)MM;
    const float mu  = gsum[col] * inv_m;
    const float var = gsumsq[col] * inv_m - mu * mu;
    const float rs  = rsqrtf(var + EPSF);
    const float g = bn_g[col], bb = bn_b[col];
#pragma unroll
    for (int j = 0; j < 8; ++j) {
        const int rw = m0 + j + 8 * h;
        const float yv = y[(size_t)rw * OC + col];
        const float v = (yv - mu) * rs * g + bb;
        out[(size_t)rw * OC + col] = gelu_exact(v) + acc[j];
    }
}

// ---------------- host launcher ----------------
extern "C" void kernel_launch(void* const* d_in, const int* in_sizes, int n_in,
                              void* d_out, int out_size, void* d_ws, size_t ws_size,
                              hipStream_t stream) {
    const float* pts    = (const float*)d_in[0];
    const float* feats  = (const float*)d_in[1];
    const float* fc1_w  = (const float*)d_in[2];
    const float* fc1_b  = (const float*)d_in[3];
    const float* ln1_g  = (const float*)d_in[4];
    const float* ln1_b  = (const float*)d_in[5];
    const float* fc2_w  = (const float*)d_in[6];
    const float* fc2_b  = (const float*)d_in[7];
    const float* ln2_g  = (const float*)d_in[8];
    const float* ln2_b  = (const float*)d_in[9];
    const float* dw_w   = (const float*)d_in[10];
    const float* dw_b   = (const float*)d_in[11];
    const float* pw_w   = (const float*)d_in[12];
    const float* pw_b   = (const float*)d_in[13];
    const float* bn_g   = (const float*)d_in[14];
    const float* bn_b   = (const float*)d_in[15];
    const float* proj_w = (const float*)d_in[16];
    const float* proj_b = (const float*)d_in[17];

    char* ws = (char*)d_ws;
    _Float16* x16     = (_Float16*)ws;  ws += (size_t)MM * IC * sizeof(_Float16);
    _Float16* feats16 = (_Float16*)ws;  ws += (size_t)MM * IC * sizeof(_Float16);
    _Float16* w1p     = (_Float16*)ws;  ws += (size_t)IC * OC * sizeof(_Float16);
    _Float16* w2p     = (_Float16*)ws;  ws += (size_t)OC * OC * sizeof(_Float16);
    _Float16* wpwp    = (_Float16*)ws;  ws += (size_t)OC * OC * sizeof(_Float16);
    _Float16* wprojp  = (_Float16*)ws;  ws += (size_t)IC * OC * sizeof(_Float16);
    float*    ybuf    = (float*)ws;     ws += (size_t)MM * OC * sizeof(float);
    float*    gacc    = (float*)ws;     ws += 256 * sizeof(float);
    float*    gsum    = gacc;
    float*    gsumsq  = gacc + OC;

    convert_zero_kernel<<<(MM * IC + 255) / 256, 256, 0, stream>>>(
        feats, fc1_w, fc2_w, pw_w, proj_w, feats16, w1p, w2p, wpwp, wprojp, gacc);

    knn_gather_kernel<<<MM / 8, 256, 0, stream>>>(pts, feats, x16);

    mlp_kernel<<<MM / 16, 256, 0, stream>>>(
        x16, w1p, fc1_b, ln1_g, ln1_b, w2p, fc2_b, ln2_g, ln2_b,
        dw_w, dw_b, wpwp, pw_b, ybuf, gsum, gsumsq);

    finalize_kernel<<<MM / 16, 256, 0, stream>>>(
        feats16, wprojp, proj_b, bn_g, bn_b, ybuf, gsum, gsumsq, (float*)d_out);
}
